// PyG_GCN_83021717831908
// MI455X (gfx1250) — matrix operand order repaired and checked by reading, not compiled, not run on hardware
//
#include <hip/hip_runtime.h>
#include <hip/hip_bf16.h>
#include <math.h>

typedef _Float16 v16h __attribute__((ext_vector_type(16)));
typedef float    v8f  __attribute__((ext_vector_type(8)));

// ---------------- atomic float add (native global_atomic_add_f32 path) ----------
__device__ __forceinline__ void atomAddF(float* p, float v) {
    unsafeAtomicAdd(p, v);   // lowers to global_atomic_add_f32 on gfx12xx
}

// ---------------- utility kernels ------------------------------------------------
__global__ void fill_f32(float* __restrict__ p, int n, float v) {
    int i = blockIdx.x * blockDim.x + threadIdx.x;
    if (i < n) p[i] = v;
}

__global__ void edge_deg(const int* __restrict__ cols, const float* __restrict__ w,
                         float* __restrict__ deg, int E) {
    int e = blockIdx.x * blockDim.x + threadIdx.x;
    if (e < E) atomAddF(&deg[cols[e]], w[e]);
}

__global__ void rsqrt_ip(float* __restrict__ d, int n) {
    int i = blockIdx.x * blockDim.x + threadIdx.x;
    if (i < n) {
        float v = d[i];                       // deg >= 1 (self loop), always > 0
        d[i] = rsqrtf(fmaxf(v, 1e-30f));
    }
}

// Pack W[K,N] (f32 row-major) into per-lane WMMA B-fragment layout (f16):
// fragment index = ((ntile*(K/32) + kstep)*32 + lane)*16 + e
// lane: col = lane&15, khalf = lane>>4 ; element e holds B[k0 + khalf*8 + (e&7) + (e>>3)*16][col]
__global__ void pack_w(const float* __restrict__ W, _Float16* __restrict__ Bp, int K, int N) {
    int tid = blockIdx.x * blockDim.x + threadIdx.x;
    if (tid >= K * N) return;
    int e     = tid & 15;
    int lane  = (tid >> 4) & 31;
    int step  = tid >> 9;
    int kst   = K >> 5;
    int nt    = step / kst;
    int k0    = (step - nt * kst) << 5;
    int col   = lane & 15;
    int kh    = lane >> 4;
    int k     = k0 + kh * 8 + (e & 7) + ((e >> 3) << 4);
    int n     = nt * 16 + col;
    Bp[tid] = (_Float16)W[(size_t)k * N + n];
}

// ---------------- WMMA GEMM: C[M,16*NT] = A[M,K] (f32 -> f16) @ Bp ---------------
// One wave per 16-row strip, NT accumulators of 16x16, K-loop of 16x16x32 WMMAs.
// A fragment: two contiguous 8-float runs per lane -> 4x global_load_b128.
template<int K, int NT>
__global__ __launch_bounds__(32)
void gemm_wmma(const float* __restrict__ A, const _Float16* __restrict__ Bp,
               float* __restrict__ C, int M) {
    constexpr int N = 16 * NT;
    const int tileM = blockIdx.x;
    const int lane  = threadIdx.x & 31;
    const int r     = lane & 15;
    const int kh    = lane >> 4;

    int grow = tileM * 16 + r;
    if (grow >= M) grow = M - 1;              // keep EXEC uniform; tail handled at store
    const float* Arow = A + (size_t)grow * K + kh * 8;

    v8f zero = {};
    v8f acc[NT];
#pragma unroll
    for (int t = 0; t < NT; ++t) acc[t] = zero;

    for (int k0 = 0; k0 < K; k0 += 32) {
        // 16-bit A layout: lane (row r, half kh) holds K = k0+kh*8+[0..8) in a[0..7]
        // and K = k0+16+kh*8+[0..8) in a[8..15]  -> 4 b128 loads, all 16B aligned.
        const float* base = Arow + k0;
        float4 q0 = *(const float4*)(base);
        float4 q1 = *(const float4*)(base + 4);
        float4 q2 = *(const float4*)(base + 16);
        float4 q3 = *(const float4*)(base + 20);
        v16h a;
        a[0]  = (_Float16)q0.x;  a[1]  = (_Float16)q0.y;
        a[2]  = (_Float16)q0.z;  a[3]  = (_Float16)q0.w;
        a[4]  = (_Float16)q1.x;  a[5]  = (_Float16)q1.y;
        a[6]  = (_Float16)q1.z;  a[7]  = (_Float16)q1.w;
        a[8]  = (_Float16)q2.x;  a[9]  = (_Float16)q2.y;
        a[10] = (_Float16)q2.z;  a[11] = (_Float16)q2.w;
        a[12] = (_Float16)q3.x;  a[13] = (_Float16)q3.y;
        a[14] = (_Float16)q3.z;  a[15] = (_Float16)q3.w;
#pragma unroll
        for (int t = 0; t < NT; ++t) {
            v16h b = *(const v16h*)(Bp + (((size_t)t * (K / 32) + (k0 >> 5)) * 32 + lane) * 16);
            acc[t] = __builtin_amdgcn_wmma_f32_16x16x32_f16(
                false, a, false, b, (short)0, acc[t], false, false);
        }
    }

    // C/D layout: VGPR v -> row (lane>>4)*8 + v, col = lane&15
    const int orow = tileM * 16 + kh * 8;
    const int ocol = lane & 15;
    if (tileM * 16 + 16 <= M) {
        // fast path: whole tile in bounds (uniform scalar branch, no exec masking)
#pragma unroll
        for (int t = 0; t < NT; ++t)
#pragma unroll
            for (int v = 0; v < 8; ++v)
                C[(size_t)(orow + v) * N + t * 16 + ocol] = acc[t][v];
    } else {
#pragma unroll
        for (int t = 0; t < NT; ++t)
#pragma unroll
            for (int v = 0; v < 8; ++v) {
                int rr = orow + v;
                if (rr < M) C[(size_t)rr * N + t * 16 + ocol] = acc[t][v];
            }
    }
}

// ---------------- edge scatter: agg[col] += XW[row] * (dinv[row]*w*dinv[col]) ----
template<int FPL>   // floats per lane; F = 32*FPL
__global__ void scatter_edges(const int* __restrict__ rows, const int* __restrict__ cols,
                              const float* __restrict__ w, const float* __restrict__ dinv,
                              const float* __restrict__ XW, float* __restrict__ agg, int E) {
    const int wpb  = blockDim.x >> 5;
    const int e    = blockIdx.x * wpb + (threadIdx.x >> 5);
    if (e >= E) return;
    const int lane = threadIdx.x & 31;
    const int rI = rows[e], cI = cols[e];
    const float nrm = dinv[rI] * w[e] * dinv[cI];
    constexpr int F = 32 * FPL;
    const float* src = XW  + (size_t)rI * F + lane * FPL;
    float*       dst = agg + (size_t)cI * F + lane * FPL;
    float v[FPL];
#pragma unroll
    for (int j = 0; j < FPL; ++j) v[j] = src[j];          // contiguous -> b64/b128 load
#pragma unroll
    for (int j = 0; j < FPL; ++j) atomAddF(dst + j, v[j] * nrm);
}

// ---------------- self loop + bias + relu (in place on agg) ----------------------
__global__ void selfloop_bias_relu(float* __restrict__ agg, const float* __restrict__ XW,
                                   const float* __restrict__ dinv, const float* __restrict__ bias,
                                   int NN, int F) {
    int idx = blockIdx.x * blockDim.x + threadIdx.x;
    if (idx >= NN * F) return;
    int i = idx / F, f = idx - i * F;
    float d = dinv[i];
    float v = agg[idx] + XW[idx] * d * d + bias[f];
    agg[idx] = fmaxf(v, 0.0f);
}

// ---------------- global add pool ------------------------------------------------
__global__ void pool_kernel(const float* __restrict__ h, const int* __restrict__ batch,
                            float* __restrict__ pooled, int NN) {
    int idx = blockIdx.x * blockDim.x + threadIdx.x;
    if (idx >= NN * 64) return;
    int i = idx >> 6, f = idx & 63;
    atomAddF(&pooled[batch[i] * 64 + f], h[idx]);
}

// ---------------- MLP head + log_softmax + argmax --------------------------------
// out layout: Y_prob[64*2] @0 | Y_hat[64] @128 | raw out[64*2] @192
__global__ void head_kernel(const float* __restrict__ pooled,
                            const float* __restrict__ l1w, const float* __restrict__ l1b,
                            const float* __restrict__ l2w, const float* __restrict__ l2b,
                            float* __restrict__ out) {
    __shared__ float hsh[64];
    __shared__ float osh[2];
    const int g = blockIdx.x, j = threadIdx.x;
    float acc = l1b[j];
    for (int k = 0; k < 64; ++k) acc += pooled[g * 64 + k] * l1w[k * 64 + j];
    hsh[j] = fmaxf(acc, 0.0f);
    __syncthreads();
    if (j < 2) {
        float o = l2b[j];
        for (int k = 0; k < 64; ++k) o += hsh[k] * l2w[k * 2 + j];
        osh[j] = o;
        out[192 + g * 2 + j] = o;
    }
    __syncthreads();
    if (j == 0) {
        float o0 = osh[0], o1 = osh[1];
        float m = fmaxf(o0, o1);
        float lse = m + logf(expf(o0 - m) + expf(o1 - m));
        out[g * 2]     = o0 - lse;
        out[g * 2 + 1] = o1 - lse;
        out[128 + g]   = (o1 > o0) ? 1.0f : 0.0f;   // argmax, ties -> 0
    }
}

// ---------------- launch ---------------------------------------------------------
extern "C" void kernel_launch(void* const* d_in, const int* in_sizes, int n_in,
                              void* d_out, int out_size, void* d_ws, size_t ws_size,
                              hipStream_t stream) {
    const float* x    = (const float*)d_in[0];
    const int*   eidx = (const int*)  d_in[1];
    const float* ew   = (const float*)d_in[2];
    const int*   bat  = (const int*)  d_in[3];
    const float* W1   = (const float*)d_in[4];
    const float* b1   = (const float*)d_in[5];
    const float* W2   = (const float*)d_in[6];
    const float* b2   = (const float*)d_in[7];
    const float* l1w  = (const float*)d_in[8];
    const float* l1b  = (const float*)d_in[9];
    const float* l2w  = (const float*)d_in[10];
    const float* l2b  = (const float*)d_in[11];
    float* out = (float*)d_out;

    const int NN = in_sizes[3];       // 100000 nodes
    const int E  = in_sizes[2];       // 1600000 edges
    const int* rows = eidx;           // edge_index[0] = source
    const int* cols = eidx + E;       // edge_index[1] = target

    // workspace carve-up (256B aligned)
    char* ws = (char*)d_ws;
    size_t o = 0;
    auto carve = [&](size_t bytes) -> void* {
        void* p = ws + o;
        o += (bytes + 255) & ~(size_t)255;
        return p;
    };
    float*    dinv   = (float*)   carve((size_t)NN * 4);            // deg -> dinv in place
    float*    B1     = (float*)   carve((size_t)NN * 128 * 4);      // XW1, then XW2
    float*    B2     = (float*)   carve((size_t)NN * 128 * 4);      // agg1 -> h1
    float*    B3     = (float*)   carve((size_t)NN * 64 * 4);       // agg2 -> h2
    float*    pooled = (float*)   carve(64 * 64 * 4);
    _Float16* W1p    = (_Float16*)carve(256 * 128 * 2);
    _Float16* W2p    = (_Float16*)carve(128 * 64 * 2);

    const int TPB = 256;
    auto blocks = [&](long long n) { return (unsigned)((n + TPB - 1) / TPB); };

    // degree + symmetric norm
    fill_f32<<<blocks(NN), TPB, 0, stream>>>(dinv, NN, 1.0f);          // self-loop weight
    edge_deg<<<blocks(E), TPB, 0, stream>>>(cols, ew, dinv, E);
    rsqrt_ip<<<blocks(NN), TPB, 0, stream>>>(dinv, NN);

    // pack weights to f16 WMMA fragments
    pack_w<<<blocks(256 * 128), TPB, 0, stream>>>(W1, W1p, 256, 128);
    pack_w<<<blocks(128 * 64),  TPB, 0, stream>>>(W2, W2p, 128, 64);

    // zero accumulators (harness poisons ws once; must re-init every call)
    fill_f32<<<blocks((long long)NN * 128), TPB, 0, stream>>>(B2, NN * 128, 0.0f);
    fill_f32<<<blocks((long long)NN * 64),  TPB, 0, stream>>>(B3, NN * 64, 0.0f);
    fill_f32<<<blocks(64 * 64), TPB, 0, stream>>>(pooled, 64 * 64, 0.0f);

    const unsigned tilesM = (unsigned)((NN + 15) / 16);

    // layer 1: GEMM -> scatter -> self-loop+bias+relu
    gemm_wmma<256, 8><<<tilesM, 32, 0, stream>>>(x, W1p, B1, NN);
    scatter_edges<4><<<(unsigned)((E + 7) / 8), 256, 0, stream>>>(rows, cols, ew, dinv, B1, B2, E);
    selfloop_bias_relu<<<blocks((long long)NN * 128), TPB, 0, stream>>>(B2, B1, dinv, b1, NN, 128);

    // layer 2
    gemm_wmma<128, 4><<<tilesM, 32, 0, stream>>>(B2, W2p, B1, NN);
    scatter_edges<2><<<(unsigned)((E + 7) / 8), 256, 0, stream>>>(rows, cols, ew, dinv, B1, B3, E);
    selfloop_bias_relu<<<blocks((long long)NN * 64), TPB, 0, stream>>>(B3, B1, dinv, b2, NN, 64);

    // pool + head
    pool_kernel<<<blocks((long long)NN * 64), TPB, 0, stream>>>(B3, bat, pooled, NN);
    head_kernel<<<64, 64, 0, stream>>>(pooled, l1w, l1b, l2w, l2b, out);
}
